// CausalSelfAttention_86723979640967
// MI455X (gfx1250) — matrix operand order repaired and checked by reading, not compiled, not run on hardware
//
#include <hip/hip_runtime.h>
#include <hip/hip_bf16.h>

// ---------------- types ----------------
typedef __attribute__((ext_vector_type(16))) _Float16 v16h;
typedef __attribute__((ext_vector_type(8)))  float    v8f;
typedef __attribute__((ext_vector_type(8)))  _Float16 h8;

#define B_   2
#define S_   4096
#define D_   512
#define H_   8
#define HD_  64
#define M_   (B_ * S_)           // 8192 rows
#define XN_  (M_ * D_)           // x elements
#define WN_  (D_ * D_)           // weight elements

__device__ __forceinline__ v8f wmma32(v16h a, v16h b, v8f c) {
  return __builtin_amdgcn_wmma_f32_16x16x32_f16(
      /*neg_a=*/false, a, /*neg_b=*/false, b,
      /*c_mod=*/(short)0, c, /*reuse_a=*/false, /*reuse_b=*/false);
}

// Async global -> LDS, 16 bytes per lane, tracked by ASYNCcnt.
__device__ __forceinline__ void async_ld_b128(unsigned lds_off, const _Float16* g) {
  asm volatile("global_load_async_to_lds_b128 %0, %1, off"
               :: "v"(lds_off), "v"(g) : "memory");
}
__device__ __forceinline__ void wait_async0() {
  asm volatile("s_wait_asynccnt 0x0" ::: "memory");
}

// A-fragment (16x32, f16): lane m = lane&15, kb = 16*(lane>>4);
// halves 0..15 = K kb..kb+15
__device__ __forceinline__ v16h load_a16(const _Float16* p0, int ld) {
  int lane = threadIdx.x & 31;
  int m    = lane & 15;
  int kb   = (lane >> 4) << 4;
  const _Float16* p = p0 + (size_t)m * ld + kb;
  union { v16h v; h8 h[2]; } u;
  u.h[0] = *(const h8*)(p);
  u.h[1] = *(const h8*)(p + 8);
  return u.v;
}

// B-fragment (32x16, f16) from "row-major N x K" storage (i.e. computing A*Bt):
// lane n = lane&15, kb = 16*(lane>>4); halves 0..15 = K kb..kb+15 of row n
__device__ __forceinline__ v16h load_bt16(const _Float16* p0, int ld) {
  int lane = threadIdx.x & 31;
  int n    = lane & 15;
  int kb   = (lane >> 4) << 4;
  const _Float16* p = p0 + (size_t)n * ld + kb;
  union { v16h v; h8 h[2]; } u;
  u.h[0] = *(const h8*)(p);
  u.h[1] = *(const h8*)(p + 8);
  return u.v;
}

// ---------------- elementwise f32 -> f16 ----------------
__global__ void cvt_kernel(const float* __restrict__ src,
                           _Float16* __restrict__ dst, int n) {
  int i      = blockIdx.x * blockDim.x + threadIdx.x;
  int stride = gridDim.x * blockDim.x;
  for (; i < n; i += stride) dst[i] = (_Float16)src[i];
}

// ---------------- fused QKV projection ----------------
// y = xh @ W^T + b, written as f16 in [B,H,S,Hd] layout. blockIdx.z picks Q/K/V.
__global__ __launch_bounds__(256) void qkv_kernel(
    const _Float16* __restrict__ xh,
    const _Float16* __restrict__ wq, const _Float16* __restrict__ wk,
    const _Float16* __restrict__ wv,
    const float* __restrict__ bq, const float* __restrict__ bk,
    const float* __restrict__ bv,
    _Float16* __restrict__ Qh, _Float16* __restrict__ Kh,
    _Float16* __restrict__ Vh) {
  const _Float16* W;
  const float*    bias;
  _Float16*       dst;
  if (blockIdx.z == 0)      { W = wq; bias = bq; dst = Qh; }
  else if (blockIdx.z == 1) { W = wk; bias = bk; dst = Kh; }
  else                      { W = wv; bias = bv; dst = Vh; }

  int wave = threadIdx.x >> 5;
  int lane = threadIdx.x & 31;
  int m0   = blockIdx.x * 128 + wave * 16;
  int n0   = blockIdx.y * 64;

  v8f acc[4] = {};
  for (int k = 0; k < D_; k += 32) {
    v16h a = load_a16(xh + (size_t)m0 * D_ + k, D_);
#pragma unroll
    for (int nt = 0; nt < 4; ++nt) {
      v16h bf = load_bt16(W + (size_t)(n0 + nt * 16) * D_ + k, D_);
      acc[nt] = wmma32(a, bf, acc[nt]);
    }
  }
  int lh = lane >> 4;
#pragma unroll
  for (int nt = 0; nt < 4; ++nt) {
#pragma unroll
    for (int r = 0; r < 8; ++r) {
      int n  = n0 + nt * 16 + (lane & 15);
      int m  = m0 + r + (lh << 3);
      int b  = m >> 12;            // /S_
      int s  = m & (S_ - 1);
      int h  = n >> 6;             // /HD_
      int hd = n & (HD_ - 1);
      float v = acc[nt][r] + bias[n];
      dst[(((size_t)(b * H_ + h)) * S_ + s) * HD_ + hd] = (_Float16)v;
    }
  }
}

// ---------------- flash attention ----------------
// grid.x = S/128 (q blocks of 128), grid.y = B*H. 8 waves, 16 q-rows per wave.
// K double-buffered via global_load_async_to_lds_b128 (ASYNCcnt);
// V double-buffered via prefetched load + transpose-scatter into LDS.
__global__ __launch_bounds__(256) void flash_kernel(
    const _Float16* __restrict__ Qh, const _Float16* __restrict__ Kh,
    const _Float16* __restrict__ Vh, const int* __restrict__ amask,
    _Float16* __restrict__ Oh) {
  __shared__ __align__(16) _Float16 ldsK[2][64 * 64];    // [buf][key][hd]
  __shared__ __align__(16) _Float16 ldsVt[2][64 * 64];   // [buf][hd][key]
  __shared__ __align__(16) _Float16 ldsP[8 * 16 * 64];   // per-wave P tiles
  __shared__ int ldsM[2][64];

  int bh   = blockIdx.y;            // b*H + h
  int b    = bh >> 3;
  int h    = bh & (H_ - 1);
  int wave = threadIdx.x >> 5;
  int lane = threadIdx.x & 31;
  int lh   = lane >> 4;
  int q0   = blockIdx.x * 128 + wave * 16;

  const _Float16* Qbase = Qh + ((size_t)bh * S_ + q0) * HD_;
  v16h qf0 = load_a16(Qbase, HD_);        // hd 0..31
  v16h qf1 = load_a16(Qbase + 32, HD_);   // hd 32..63

  v8f   o[4] = {};
  float mrow[8], lrow[8];
#pragma unroll
  for (int r = 0; r < 8; ++r) { mrow[r] = -1e38f; lrow[r] = 0.f; }

  // stage K async + V transpose-scatter + pad mask for key block kb into buf
  auto stage = [&](int kb, int buf) {
    int t   = threadIdx.x;
    int row = t >> 2;                     // key row 0..63
    int col = (t & 3) * 16;               // hd offset
    const _Float16* kg = Kh + ((size_t)bh * S_ + (size_t)kb * 64 + row) * HD_ + col;
    unsigned kdst = (unsigned)(uintptr_t)&ldsK[buf][row * 64 + col];
    async_ld_b128(kdst,       kg);
    async_ld_b128(kdst + 16u, kg + 8);
    const _Float16* vg = Vh + ((size_t)bh * S_ + (size_t)kb * 64 + row) * HD_ + col;
    h8 v0 = *(const h8*)vg;
    h8 v1 = *(const h8*)(vg + 8);
#pragma unroll
    for (int i = 0; i < 8; ++i) ldsVt[buf][(col + i) * 64 + row]     = v0[i];
#pragma unroll
    for (int i = 0; i < 8; ++i) ldsVt[buf][(col + 8 + i) * 64 + row] = v1[i];
    if (t < 64) ldsM[buf][t] = amask[b * S_ + kb * 64 + t];
  };

  const int nkb = blockIdx.x * 2 + 2;     // causal: keys up to q0_block+127
  stage(0, 0);

  for (int kb = 0; kb < nkb; ++kb) {
    int cur = kb & 1;
    wait_async0();                        // this wave's K(cur) transfers done
    __syncthreads();                      // all waves' staging of cur done;
                                          // buf (1-cur) readers from kb-1 done
    if (kb + 1 < nkb) stage(kb + 1, cur ^ 1);   // prefetch overlaps compute

    // ---- S = (Q K^T) * scale, 16x64 per wave ----
    v8f s[4];
#pragma unroll
    for (int nt = 0; nt < 4; ++nt) {
      v8f acc = {};
      acc = wmma32(qf0, load_bt16(&ldsK[cur][nt * 16 * 64], 64), acc);
      acc = wmma32(qf1, load_bt16(&ldsK[cur][nt * 16 * 64 + 32], 64), acc);
      s[nt] = acc;
    }

    // ---- mask + online softmax (fp32, C-layout) ----
    int kcol  = (lane & 15);
    int rbase = q0 + (lh << 3);
    float mnew[8];
#pragma unroll
    for (int r = 0; r < 8; ++r) {
      int   qi = rbase + r;
      float mx = mrow[r];
#pragma unroll
      for (int nt = 0; nt < 4; ++nt) {
        int   kloc = nt * 16 + kcol;
        int   ki   = kb * 64 + kloc;
        float v    = s[nt][r] * 0.125f;        // 1/sqrt(64)
        if (ki > qi || ldsM[cur][kloc] == 0) v = -1e38f;
        s[nt][r] = v;
        mx = fmaxf(mx, v);
      }
#pragma unroll
      for (int off = 1; off < 16; off <<= 1)
        mx = fmaxf(mx, __shfl_xor(mx, off, 32));
      mnew[r] = mx;
    }

    float alpha[8];
#pragma unroll
    for (int r = 0; r < 8; ++r)
      alpha[r] = (mrow[r] <= -1e30f) ? 0.f : __expf(mrow[r] - mnew[r]);

#pragma unroll
    for (int r = 0; r < 8; ++r) {
      float rs = 0.f;
#pragma unroll
      for (int nt = 0; nt < 4; ++nt) {
        float v = s[nt][r];
        float p = (v <= -1e30f) ? 0.f : __expf(v - mnew[r]);
        s[nt][r] = p;
        rs += p;
      }
#pragma unroll
      for (int off = 1; off < 16; off <<= 1) rs += __shfl_xor(rs, off, 32);
      lrow[r] = alpha[r] * lrow[r] + rs;
      mrow[r] = mnew[r];
    }

#pragma unroll
    for (int nt = 0; nt < 4; ++nt)
#pragma unroll
      for (int r = 0; r < 8; ++r) o[nt][r] *= alpha[r];

    // ---- P: C-layout -> LDS (row-major 16x64 per wave) -> A-fragments ----
    _Float16* pw = &ldsP[wave * 16 * 64];
#pragma unroll
    for (int nt = 0; nt < 4; ++nt)
#pragma unroll
      for (int r = 0; r < 8; ++r)
        pw[(r + (lh << 3)) * 64 + nt * 16 + kcol] = (_Float16)s[nt][r];
    __builtin_amdgcn_wave_barrier();   // LDS ops are in-order per-wave

    // ---- O += P * V ----
#pragma unroll
    for (int ks = 0; ks < 2; ++ks) {
      v16h pa = load_a16(pw + ks * 32, 64);
#pragma unroll
      for (int nt = 0; nt < 4; ++nt) {
        v16h vb = load_bt16(&ldsVt[cur][(nt * 16) * 64 + ks * 32], 64);
        o[nt] = wmma32(pa, vb, o[nt]);
      }
    }
  }

  // ---- normalize + store to Oh [B,S,D] ----
  float inv[8];
#pragma unroll
  for (int r = 0; r < 8; ++r) inv[r] = (lrow[r] > 0.f) ? 1.f / lrow[r] : 0.f;
#pragma unroll
  for (int nt = 0; nt < 4; ++nt)
#pragma unroll
    for (int r = 0; r < 8; ++r) {
      int q   = q0 + r + (lh << 3);
      int col = h * HD_ + nt * 16 + (lane & 15);
      Oh[((size_t)(b * S_ + q)) * D_ + col] = (_Float16)(o[nt][r] * inv[r]);
    }
}

// ---------------- output projection ----------------
__global__ __launch_bounds__(256) void oproj_kernel(
    const _Float16* __restrict__ Oh, const _Float16* __restrict__ wo,
    const float* __restrict__ bo, float* __restrict__ out) {
  int wave = threadIdx.x >> 5;
  int lane = threadIdx.x & 31;
  int m0   = blockIdx.x * 128 + wave * 16;
  int n0   = blockIdx.y * 64;

  v8f acc[4] = {};
  for (int k = 0; k < D_; k += 32) {
    v16h a = load_a16(Oh + (size_t)m0 * D_ + k, D_);
#pragma unroll
    for (int nt = 0; nt < 4; ++nt) {
      v16h bf = load_bt16(wo + (size_t)(n0 + nt * 16) * D_ + k, D_);
      acc[nt] = wmma32(a, bf, acc[nt]);
    }
  }
  int lh = lane >> 4;
#pragma unroll
  for (int nt = 0; nt < 4; ++nt)
#pragma unroll
    for (int r = 0; r < 8; ++r) {
      int n = n0 + nt * 16 + (lane & 15);
      int m = m0 + r + (lh << 3);
      out[(size_t)m * D_ + n] = acc[nt][r] + bo[n];
    }
}

// ---------------- launcher ----------------
extern "C" void kernel_launch(void* const* d_in, const int* in_sizes, int n_in,
                              void* d_out, int out_size, void* d_ws, size_t ws_size,
                              hipStream_t stream) {
  const float* x  = (const float*)d_in[0];
  const int*   am = (const int*)d_in[1];
  const float* Wq = (const float*)d_in[2];
  const float* bq = (const float*)d_in[3];
  const float* Wk = (const float*)d_in[4];
  const float* bk = (const float*)d_in[5];
  const float* Wv = (const float*)d_in[6];
  const float* bv = (const float*)d_in[7];
  const float* Wo = (const float*)d_in[8];
  const float* bo = (const float*)d_in[9];
  float* out = (float*)d_out;

  _Float16* w   = (_Float16*)d_ws;
  _Float16* xh  = w;
  _Float16* wqh = xh + (size_t)XN_;
  _Float16* wkh = wqh + (size_t)WN_;
  _Float16* wvh = wkh + (size_t)WN_;
  _Float16* woh = wvh + (size_t)WN_;
  _Float16* Qh  = woh + (size_t)WN_;
  _Float16* Kh  = Qh + (size_t)XN_;
  _Float16* Vh  = Kh + (size_t)XN_;
  _Float16* Oh  = Vh + (size_t)XN_;

  // 1) fp32 -> f16 conversions
  cvt_kernel<<<2048, 256, 0, stream>>>(x,  xh,  XN_);
  cvt_kernel<<<512,  256, 0, stream>>>(Wq, wqh, WN_);
  cvt_kernel<<<512,  256, 0, stream>>>(Wk, wkh, WN_);
  cvt_kernel<<<512,  256, 0, stream>>>(Wv, wvh, WN_);
  cvt_kernel<<<512,  256, 0, stream>>>(Wo, woh, WN_);

  // 2) fused QKV projection (z selects Q/K/V)
  dim3 gq(M_ / 128, D_ / 64, 3);
  qkv_kernel<<<gq, 256, 0, stream>>>(xh, wqh, wkh, wvh, bq, bk, bv, Qh, Kh, Vh);

  // 3) flash attention
  dim3 ga(S_ / 128, B_ * H_, 1);
  flash_kernel<<<ga, 256, 0, stream>>>(Qh, Kh, Vh, am, Oh);

  // 4) output projection
  dim3 go(M_ / 128, D_ / 64, 1);
  oproj_kernel<<<go, 256, 0, stream>>>(Oh, woh, bo, out);
}
